// LeNetNoisyBatch_38500086842157
// MI455X (gfx1250) — compile-verified, matrix operand order checked by reading, not run on hardware
//
#include <hip/hip_runtime.h>

// ---------------------------------------------------------------------------
// CDNA5 (gfx1250) bf16-WMMA noisy 3-layer MLP.
// Round 4: use __builtin_amdgcn_sched_group_barrier to force a
// DS-read / WMMA software pipeline (B fragment nt+1 in flight during WMMA nt)
// instead of the scheduler's serialized load->wait->wmma clustering.
// ---------------------------------------------------------------------------

typedef __attribute__((ext_vector_type(16))) __bf16 v16bf;
typedef __attribute__((ext_vector_type(8)))  __bf16 v8bf;
typedef __attribute__((ext_vector_type(8)))  float  v8f;
typedef __attribute__((ext_vector_type(4)))  float  v4f;

#define LDSK 40   // LDS row stride in bf16 (80B -> conflict-free ds_load_b128)

__device__ __forceinline__ v16bf load_frag(const __bf16* tile, int row_base, int lane) {
  const int m  = lane & 15;
  const int kh = (lane >> 4) * 8;
  const __bf16* p = tile + (row_base + m) * LDSK + kh;
  union { v16bf v; v8bf h[2]; } u;
  u.h[0] = *(const v8bf*)(p);
  u.h[1] = *(const v8bf*)(p + 16);
  return u.v;
}

__device__ __forceinline__ v8f wmma_bf16(v16bf a, v16bf b, v8f c) {
  return __builtin_amdgcn_wmma_f32_16x16x32_bf16(false, a, false, b, (short)0, c,
                                                 false, false);
}

// WMMA sweep over NT n-tiles with scheduler-enforced pipelining:
//   [6 DS reads] ([1 WMMA][2 DS reads]) x (NT-2) [2 WMMA]
// -> every B fragment's ds_loads issue one full WMMA before their use.
template <int NT>
__device__ __forceinline__ void wmma_sweep(const __bf16* Wt, const __bf16* At,
                                           int arow0, v8f* acc, int lane) {
  const v16bf a = load_frag(At, arow0, lane);
#pragma unroll
  for (int nt = 0; nt < NT; ++nt) {
    const v16bf b = load_frag(Wt, nt * 16, lane);
    acc[nt] = wmma_bf16(a, b, acc[nt]);
  }
  if constexpr (NT >= 3) {
    __builtin_amdgcn_sched_group_barrier(0x100, 6, 0);  // DS read: A + B0 + B1
#pragma unroll
    for (int i = 0; i < NT - 2; ++i) {
      __builtin_amdgcn_sched_group_barrier(0x008, 1, 0);  // 1 WMMA
      __builtin_amdgcn_sched_group_barrier(0x100, 2, 0);  // next B frag loads
    }
    __builtin_amdgcn_sched_group_barrier(0x008, 2, 0);    // last 2 WMMAs
  }
}

// W-tile prefetch / store. NITEMS = 2*rows (two 16-elem halves per 32-wide row).
template <int NITEMS, int KP>
__device__ __forceinline__ void prefetch_w(const __bf16* __restrict__ Wbf, int kb,
                                           int tid, v8bf (*wpf)[2]) {
#pragma unroll
  for (int c = 0; c < (NITEMS + 255) / 256; ++c) {
    const int i = tid + 256 * c;
    if (i < NITEMS) {
      const int row = i >> 1, half = (i & 1) * 16;
      const __bf16* src = Wbf + (size_t)row * KP + kb + half;
      wpf[c][0] = *(const v8bf*)(src);
      wpf[c][1] = *(const v8bf*)(src + 8);
    }
  }
}
template <int NITEMS>
__device__ __forceinline__ void store_w(__bf16* Wt, int tid, const v8bf (*wpf)[2]) {
#pragma unroll
  for (int c = 0; c < (NITEMS + 255) / 256; ++c) {
    const int i = tid + 256 * c;
    if (i < NITEMS) {
      const int row = i >> 1, half = (i & 1) * 16;
      __bf16* dst = Wt + row * LDSK + half;
      *(v8bf*)(dst)     = wpf[c][0];
      *(v8bf*)(dst + 8) = wpf[c][1];
    }
  }
}

// ---------------------------------------------------------------------------
// prep: convert + zero-pad weights to bf16. W1bf 320x800, W2bf 112x320, W3bf 16x128
// ---------------------------------------------------------------------------
__global__ __launch_bounds__(256) void prep_kernel(
    const float* __restrict__ W1, const float* __restrict__ W2,
    const float* __restrict__ W3, __bf16* __restrict__ W1bf,
    __bf16* __restrict__ W2bf, __bf16* __restrict__ W3bf) {
  const int idx = blockIdx.x * 256 + threadIdx.x;
  if (idx < 320 * 800) {
    const int n = idx / 800, k = idx % 800;
    const float v = (n < 300 && k < 784) ? W1[n * 784 + k] : 0.f;
    W1bf[idx] = (__bf16)v;
  } else if (idx < 320 * 800 + 112 * 320) {
    const int j = idx - 320 * 800;
    const int n = j / 320, k = j % 320;
    const float v = (n < 100 && k < 300) ? W2[n * 300 + k] : 0.f;
    W2bf[j] = (__bf16)v;
  } else if (idx < 320 * 800 + 112 * 320 + 16 * 128) {
    const int j = idx - (320 * 800 + 112 * 320);
    const int n = j / 128, k = j % 128;
    const float v = (n < 10 && k < 100) ? W3[n * 100 + k] : 0.f;
    W3bf[j] = (__bf16)v;
  }
}

// ---------------------------------------------------------------------------
// gemm1: h1 = relu(x @ W1^T + b1) + per-block BN partials.
// Grid 512(M) x 2(N): block tile 128(M) x 160(N), K 784 -> 800 (25 steps).
// ---------------------------------------------------------------------------
__global__ __launch_bounds__(256, 1) void gemm1_kernel(
    const float* __restrict__ x, const __bf16* __restrict__ Wbf,
    const float* __restrict__ bias, float* __restrict__ h1,
    float* __restrict__ part) {
  __shared__ __bf16 At[128 * LDSK];
  __shared__ __bf16 Wt[160 * LDSK];
  __shared__ float lsum[160];
  __shared__ float lssq[160];

  const int tid   = threadIdx.x;
  const int lane  = tid & 31;
  const int wave  = tid >> 5;
  const int mblk  = blockIdx.x >> 1;
  const int nb    = blockIdx.x & 1;
  const int m0    = mblk * 128;
  const int nrow0 = nb * 160;  // this block's W row / output column base
  const __bf16* Wbase = Wbf + (size_t)nrow0 * 800;

  if (tid < 160) { lsum[tid] = 0.f; lssq[tid] = 0.f; }

  const v8f vzero = {0.f, 0.f, 0.f, 0.f, 0.f, 0.f, 0.f, 0.f};
  v8f acc[10];
#pragma unroll
  for (int i = 0; i < 10; ++i) acc[i] = vzero;

  const int arow  = tid >> 1;
  const int ahalf = (tid & 1) * 16;
  const float* arowp = x + (size_t)(m0 + arow) * 784;

  v4f apf[4];        // next A half-row (x rows are 16B aligned)
  v8bf wpf[2][2];    // next W tile (320 half-rows over 256 threads)

  {
    const int kc = ahalf;
#pragma unroll
    for (int c = 0; c < 4; ++c) apf[c] = *(const v4f*)(arowp + kc + 4 * c);
  }
  prefetch_w<320, 800>(Wbase, 0, tid, wpf);

  for (int kt = 0; kt < 25; ++kt) {
    const int kb = kt * 32;
    __syncthreads();
    {  // A tile: f32 -> bf16 into LDS (from prefetched regs)
      union { v8bf v[2]; __bf16 e[16]; } u;
#pragma unroll
      for (int c = 0; c < 4; ++c)
#pragma unroll
        for (int j = 0; j < 4; ++j) u.e[4 * c + j] = (__bf16)apf[c][j];
      __bf16* dst = At + arow * LDSK + ahalf;
      *(v8bf*)(dst)     = u.v[0];
      *(v8bf*)(dst + 8) = u.v[1];
    }
    store_w<320>(Wt, tid, wpf);
    __syncthreads();
    if (kt + 1 < 25) {  // issue next k-step's global loads before compute
      const int kc = kb + 32 + ahalf;
      if (kc < 784) {
#pragma unroll
        for (int c = 0; c < 4; ++c) apf[c] = *(const v4f*)(arowp + kc + 4 * c);
      } else {
#pragma unroll
        for (int c = 0; c < 4; ++c) apf[c] = (v4f){0.f, 0.f, 0.f, 0.f};
      }
      prefetch_w<320, 800>(Wbase, kb + 32, tid, wpf);
    }
    wmma_sweep<10>(Wt, At, wave * 16, acc, lane);
  }

  // Epilogue: bias + relu, store h1, per-column partials via LDS ds atomics.
  const int col0  = lane & 15;
  const int rbase = m0 + wave * 16 + ((lane >> 4) * 8);
#pragma unroll
  for (int nt = 0; nt < 10; ++nt) {
    const int col = nrow0 + nt * 16 + col0;
    if (col < 300) {
      const float bv = bias[col];
      float s = 0.f, sq = 0.f;
#pragma unroll
      for (int r = 0; r < 8; ++r) {
        float v = acc[nt][r] + bv;
        v = v > 0.f ? v : 0.f;
        h1[(size_t)(rbase + r) * 300 + col] = v;
        s += v;
        sq += v * v;
      }
      atomicAdd(&lsum[col - nrow0], s);
      atomicAdd(&lssq[col - nrow0], sq);
    }
  }
  __syncthreads();
  if (tid < 160) {  // deterministic per-block partials (disjoint column slices)
    part[(size_t)(2 * mblk) * 320 + nrow0 + tid]     = lsum[tid];
    part[(size_t)(2 * mblk + 1) * 320 + nrow0 + tid] = lssq[tid];
  }
}

// ---------------------------------------------------------------------------
// finalize: reduce block partials -> per-column affine A = g*rstd, B = be - A*mean.
// ---------------------------------------------------------------------------
__global__ void finalize_kernel(const float* __restrict__ part, int stride,
                                int nblocks, const float* __restrict__ g,
                                const float* __restrict__ be,
                                float* __restrict__ A, float* __restrict__ Bc,
                                int n) {
  const int i = blockIdx.x * blockDim.x + threadIdx.x;
  if (i >= n) return;
  float s = 0.f, sq = 0.f;
  for (int b = 0; b < nblocks; ++b) {
    s  += part[(size_t)(2 * b) * stride + i];
    sq += part[(size_t)(2 * b + 1) * stride + i];
  }
  const float inv  = 1.0f / 65536.0f;
  const float mean = s * inv;
  const float var  = sq * inv - mean * mean;
  const float rstd = rsqrtf(var + 1e-5f);
  const float a    = g[i] * rstd;
  A[i]  = a;
  Bc[i] = be[i] - a * mean;
}

// ---------------------------------------------------------------------------
// gemm2: h2 = relu((s1*(A1*h1+B1)+m1) @ W2^T + b2) + BN2 partials.
// K 300 -> 320 (10 steps), N 100 -> 112 (7 tiles).
// ---------------------------------------------------------------------------
__global__ __launch_bounds__(256, 1) void gemm2_kernel(
    const float* __restrict__ h1, const float* __restrict__ s1,
    const float* __restrict__ m1, const __bf16* __restrict__ Wbf,
    const float* __restrict__ bias, const float* __restrict__ Ag,
    const float* __restrict__ Bg, float* __restrict__ h2,
    float* __restrict__ part) {
  __shared__ __bf16 At[128 * LDSK];
  __shared__ __bf16 Wt[112 * LDSK];
  __shared__ float lA[320];
  __shared__ float lB[320];
  __shared__ float lsum[112];
  __shared__ float lssq[112];

  const int tid  = threadIdx.x;
  const int lane = tid & 31;
  const int wave = tid >> 5;
  const int m0   = blockIdx.x * 128;

  if (tid < 320) {
    lA[tid] = (tid < 300) ? Ag[tid] : 0.f;
    lB[tid] = (tid < 300) ? Bg[tid] : 0.f;
  }
  if (tid < 112) { lsum[tid] = 0.f; lssq[tid] = 0.f; }

  const v8f vzero = {0.f, 0.f, 0.f, 0.f, 0.f, 0.f, 0.f, 0.f};
  v8f acc[7];
#pragma unroll
  for (int i = 0; i < 7; ++i) acc[i] = vzero;

  const int arow  = tid >> 1;
  const int ahalf = (tid & 1) * 16;
  const size_t gr = (size_t)(m0 + arow) * 300;

  float hp[16], sp[16], mp[16];  // prefetched h1/s1/m1 (scalar loads, zero past k = 300)
  v8bf wpf[1][2];

#pragma unroll
  for (int j = 0; j < 16; ++j) {
    const int k = ahalf + j;
    const bool ok = (k < 300);
    hp[j] = ok ? h1[gr + k] : 0.f;
    sp[j] = ok ? s1[gr + k] : 0.f;
    mp[j] = ok ? m1[gr + k] : 0.f;
  }
  prefetch_w<224, 320>(Wbf, 0, tid, wpf);

  for (int kt = 0; kt < 10; ++kt) {
    const int kb = kt * 32;
    __syncthreads();
    {  // fused BN1-apply + noise1 + cvt into A tile
      union { v8bf v[2]; __bf16 e[16]; } u;
#pragma unroll
      for (int j = 0; j < 16; ++j) {
        const int k = kb + ahalf + j;
        float v = 0.f;
        if (k < 300) v = fmaf(sp[j], fmaf(lA[k], hp[j], lB[k]), mp[j]);
        u.e[j] = (__bf16)v;
      }
      __bf16* dst = At + arow * LDSK + ahalf;
      *(v8bf*)(dst)     = u.v[0];
      *(v8bf*)(dst + 8) = u.v[1];
    }
    store_w<224>(Wt, tid, wpf);
    __syncthreads();
    if (kt + 1 < 10) {
#pragma unroll
      for (int j = 0; j < 16; ++j) {
        const int k = kb + 32 + ahalf + j;
        const bool ok = (k < 300);
        hp[j] = ok ? h1[gr + k] : 0.f;
        sp[j] = ok ? s1[gr + k] : 0.f;
        mp[j] = ok ? m1[gr + k] : 0.f;
      }
      prefetch_w<224, 320>(Wbf, kb + 32, tid, wpf);
    }
    wmma_sweep<7>(Wt, At, wave * 16, acc, lane);
  }

  const int col0  = lane & 15;
  const int rbase = m0 + wave * 16 + ((lane >> 4) * 8);
#pragma unroll
  for (int nt = 0; nt < 7; ++nt) {
    const int col = nt * 16 + col0;
    if (col < 100) {
      const float bv = bias[col];
      float s = 0.f, sq = 0.f;
#pragma unroll
      for (int r = 0; r < 8; ++r) {
        float v = acc[nt][r] + bv;
        v = v > 0.f ? v : 0.f;
        h2[(size_t)(rbase + r) * 100 + col] = v;
        s += v;
        sq += v * v;
      }
      atomicAdd(&lsum[col], s);
      atomicAdd(&lssq[col], sq);
    }
  }
  __syncthreads();
  if (tid < 112) {
    part[(size_t)(2 * blockIdx.x) * 112 + tid]     = lsum[tid];
    part[(size_t)(2 * blockIdx.x + 1) * 112 + tid] = lssq[tid];
  }
}

// ---------------------------------------------------------------------------
// gemm3: out = (s2*(A2*h2+B2)+m2) @ W3^T + b3.  K 100 -> 128, N 10 -> 16.
// ---------------------------------------------------------------------------
__global__ __launch_bounds__(256) void gemm3_kernel(
    const float* __restrict__ h2, const float* __restrict__ s2,
    const float* __restrict__ m2, const __bf16* __restrict__ Wbf,
    const float* __restrict__ bias, const float* __restrict__ Ag,
    const float* __restrict__ Bg, float* __restrict__ out) {
  __shared__ __bf16 At[128 * LDSK];
  __shared__ __bf16 Wt[16 * LDSK];
  __shared__ float lA[128];
  __shared__ float lB[128];

  const int tid  = threadIdx.x;
  const int lane = tid & 31;
  const int wave = tid >> 5;
  const int m0   = blockIdx.x * 128;

  if (tid < 128) {
    lA[tid] = (tid < 100) ? Ag[tid] : 0.f;
    lB[tid] = (tid < 100) ? Bg[tid] : 0.f;
  }

  v8f acc[1] = {{0.f, 0.f, 0.f, 0.f, 0.f, 0.f, 0.f, 0.f}};

  const int arow  = tid >> 1;
  const int ahalf = (tid & 1) * 16;
  const size_t gr = (size_t)(m0 + arow) * 100;

  float hp[16], sp[16], mp[16];
  v8bf wpf[1][2];

#pragma unroll
  for (int j = 0; j < 16; ++j) {
    const int k = ahalf + j;
    const bool ok = (k < 100);
    hp[j] = ok ? h2[gr + k] : 0.f;
    sp[j] = ok ? s2[gr + k] : 0.f;
    mp[j] = ok ? m2[gr + k] : 0.f;
  }
  prefetch_w<32, 128>(Wbf, 0, tid, wpf);

  for (int kt = 0; kt < 4; ++kt) {
    const int kb = kt * 32;
    __syncthreads();
    {  // fused BN2-apply + noise2 into A tile
      union { v8bf v[2]; __bf16 e[16]; } u;
#pragma unroll
      for (int j = 0; j < 16; ++j) {
        const int k = kb + ahalf + j;
        float v = 0.f;
        if (k < 100) v = fmaf(sp[j], fmaf(lA[k], hp[j], lB[k]), mp[j]);
        u.e[j] = (__bf16)v;
      }
      __bf16* dst = At + arow * LDSK + ahalf;
      *(v8bf*)(dst)     = u.v[0];
      *(v8bf*)(dst + 8) = u.v[1];
    }
    store_w<32>(Wt, tid, wpf);
    __syncthreads();
    if (kt + 1 < 4) {
#pragma unroll
      for (int j = 0; j < 16; ++j) {
        const int k = kb + 32 + ahalf + j;
        const bool ok = (k < 100);
        hp[j] = ok ? h2[gr + k] : 0.f;
        sp[j] = ok ? s2[gr + k] : 0.f;
        mp[j] = ok ? m2[gr + k] : 0.f;
      }
      prefetch_w<32, 128>(Wbf, kb + 32, tid, wpf);
    }
    wmma_sweep<1>(Wt, At, wave * 16, acc, lane);
  }

  const int col   = lane & 15;
  const int rbase = m0 + wave * 16 + ((lane >> 4) * 8);
  if (col < 10) {
    const float bv = bias[col];
#pragma unroll
    for (int r = 0; r < 8; ++r)
      out[(size_t)(rbase + r) * 10 + col] = acc[0][r] + bv;
  }
}

// ---------------------------------------------------------------------------
extern "C" void kernel_launch(void* const* d_in, const int* in_sizes, int n_in,
                              void* d_out, int out_size, void* d_ws,
                              size_t ws_size, hipStream_t stream) {
  const float* x   = (const float*)d_in[0];
  const float* W1  = (const float*)d_in[1];
  const float* b1  = (const float*)d_in[2];
  const float* g1  = (const float*)d_in[3];
  const float* be1 = (const float*)d_in[4];
  const float* s1  = (const float*)d_in[5];
  const float* m1  = (const float*)d_in[6];
  const float* W2  = (const float*)d_in[7];
  const float* b2  = (const float*)d_in[8];
  const float* g2  = (const float*)d_in[9];
  const float* be2 = (const float*)d_in[10];
  const float* s2  = (const float*)d_in[11];
  const float* m2  = (const float*)d_in[12];
  const float* W3  = (const float*)d_in[13];
  const float* b3  = (const float*)d_in[14];
  float* out = (float*)d_out;

  char* ws   = (char*)d_ws;
  size_t off = 0;
  auto take = [&](size_t bytes) -> void* {
    void* p = (void*)(ws + off);
    off += (bytes + 255) & ~(size_t)255;
    return p;
  };
  __bf16* W1bf = (__bf16*)take((size_t)320 * 800 * 2);
  __bf16* W2bf = (__bf16*)take((size_t)112 * 320 * 2);
  __bf16* W3bf = (__bf16*)take((size_t)16 * 128 * 2);
  float* part1 = (float*)take((size_t)2 * 512 * 320 * 4);
  float* part2 = (float*)take((size_t)2 * 512 * 112 * 4);
  float* A1    = (float*)take(320 * 4);
  float* B1    = (float*)take(320 * 4);
  float* A2    = (float*)take(112 * 4);
  float* B2    = (float*)take(112 * 4);
  float* h1    = (float*)take((size_t)65536 * 300 * 4);
  float* h2    = (float*)take((size_t)65536 * 100 * 4);
  (void)ws_size; (void)n_in; (void)in_sizes; (void)out_size;

  prep_kernel<<<1148, 256, 0, stream>>>(W1, W2, W3, W1bf, W2bf, W3bf);
  gemm1_kernel<<<1024, 256, 0, stream>>>(x, W1bf, b1, h1, part1);
  finalize_kernel<<<2, 256, 0, stream>>>(part1, 320, 512, g1, be1, A1, B1, 300);
  gemm2_kernel<<<512, 256, 0, stream>>>(h1, s1, m1, W2bf, b2, A1, B1, h2, part2);
  finalize_kernel<<<1, 128, 0, stream>>>(part2, 112, 512, g2, be2, A2, B2, 100);
  gemm3_kernel<<<512, 256, 0, stream>>>(h2, s2, m2, W3bf, b3, A2, B2, out);
}
